// TransformerModel_44813688767054
// MI455X (gfx1250) — compile-verified, matrix operand order checked by reading, not run on hardware
//
#include <hip/hip_runtime.h>
#include <hip/hip_bf16.h>
#include <math.h>

// ---------------------------------------------------------------------------
// GPT-style transformer forward for gfx1250 (MI455X).
// HBM-bound (~290 GFLOP vs ~0.5 GB @ 23.3 TB/s) =>
//  - fp32 weights read ~once from HBM (B tiles re-hit the 192MB L2),
//  - compute via v_wmma_f32_16x16x32_bf16 (fp32 accumulate),
//  - A tiles staged LDS-side by the Tensor Data Mover (tensor_load_to_lds,
//    with D# pad fields generating the LDS bank-padding), tracked by TENSORcnt,
//  - B tiles register-transposed 4x4 -> packed bf16 -> ds_store_b64,
//  - all GEMM epilogues (bias/residual/relu/bf16-cast) fused.
// ---------------------------------------------------------------------------

typedef __attribute__((ext_vector_type(16))) __bf16        v16bf;
typedef __attribute__((ext_vector_type(8)))  float         v8f;
typedef __attribute__((ext_vector_type(4)))  unsigned int  v4u;
typedef __attribute__((ext_vector_type(8)))  int           v8i;
typedef __attribute__((ext_vector_type(4)))  int           v4i;

static __device__ __forceinline__ unsigned pack_bf2(float a, float b) {
    union { __hip_bfloat16 h; unsigned short s; } ua, ub;
    ua.h = __float2bfloat16(a);
    ub.h = __float2bfloat16(b);
    return (unsigned)ua.s | ((unsigned)ub.s << 16);
}

// ---------------------------------------------------------------------------
// TDM: 2D tile load Global -> LDS with optional LDS padding.
// D# packing per CDNA5 ISA (group0) / §8.4 (group1).
// All arguments must be wave-uniform. 2B elements (data_size code 1).
// This toolchain exposes the 6-arg builtin:
//   (uint32x4 g0, int32x8 g1, int32x4 g2, int32x4 g3, int32x8 extra, i32 cpol)
// ---------------------------------------------------------------------------
static __device__ __forceinline__ void tdm_load_2d_bf16(
    unsigned lds_addr, const void* gaddr,
    unsigned tile_d0, unsigned tile_d1,   // elements
    unsigned long long stride0,           // elements between rows
    unsigned pad_interval_code, unsigned pad_amount_code, bool pad_en) {
    unsigned long long ga = (unsigned long long)gaddr;
    v4u g0;
    g0.x = 0x1u;                                   // count=1, user descriptor
    g0.y = lds_addr;                               // bits 63:32
    g0.z = (unsigned)ga;                           // global_addr[31:0]
    g0.w = (unsigned)((ga >> 32) & 0x01FFFFFFu)    // global_addr[56:32]
         | (2u << 30);                             // type=2 ("image")
    v8i g1;
    unsigned w0 = (1u << 16)                                   // data_size=2B
                | (pad_en ? (1u << 20) : 0u)                   // pad_enable
                | ((pad_interval_code & 7u) << 22)
                | ((pad_amount_code & 127u) << 25);
    // tensor dims == tile dims (tile fully in-range; no OOB clipping needed)
    unsigned td0 = tile_d0, td1 = tile_d1;
    g1[0] = (int)w0;
    g1[1] = (int)((td0 & 0xFFFFu) << 16);                  // tensor_dim0[15:0]
    g1[2] = (int)((td0 >> 16) | ((td1 & 0xFFFFu) << 16));  // d0 hi | d1 lo
    g1[3] = (int)((td1 >> 16) | ((tile_d0 & 0xFFFFu) << 16));  // d1 hi|tile0
    g1[4] = (int)(tile_d1 & 0xFFFFu);                      // tile1 | tile2=0
    g1[5] = (int)(unsigned)(stride0 & 0xFFFFFFFFull);      // dim0_stride lo
    g1[6] = (int)(unsigned)((stride0 >> 32) & 0xFFFFull);  // dim0_stride hi
    g1[7] = 0;
    v4i z4 = {0, 0, 0, 0};
    v8i z8 = {0, 0, 0, 0, 0, 0, 0, 0};
    __builtin_amdgcn_tensor_load_to_lds(g0, g1, z4, z4, z8, 0);
}

// ---------------------------------------------------------------------------
// Embedding: x[b,t,:] = tok_emb[idx[b,t],:] + pos_emb[t,:]
// ---------------------------------------------------------------------------
__global__ void embed_kernel(const int* __restrict__ idx,
                             const float* __restrict__ tok,
                             const float* __restrict__ pos,
                             float* __restrict__ x,
                             int Tv, int C, long total) {
    long gid = (long)blockIdx.x * blockDim.x + threadIdx.x;
    if (gid >= total) return;
    int c     = (int)(gid % C);
    int token = (int)(gid / C);
    int tpos  = token % Tv;
    int tk    = idx[token];
    x[gid] = tok[(size_t)tk * C + c] + pos[(size_t)tpos * C + c];
}

// ---------------------------------------------------------------------------
// LayerNorm over C, emits bf16. One block per token.
// ---------------------------------------------------------------------------
__global__ __launch_bounds__(256)
void layernorm_bf16_kernel(const float* __restrict__ x,
                           const float* __restrict__ g,
                           const float* __restrict__ b,
                           __hip_bfloat16* __restrict__ out, int C) {
    const int t = blockIdx.x;
    const float* xr = x + (size_t)t * C;
    __shared__ float red[256];
    const int tid = threadIdx.x;

    float s = 0.f;
    for (int c = tid; c < C; c += 256) s += xr[c];
    red[tid] = s; __syncthreads();
    for (int o = 128; o > 0; o >>= 1) {
        if (tid < o) red[tid] += red[tid + o];
        __syncthreads();
    }
    const float mean = red[0] / (float)C;
    __syncthreads();

    float sq = 0.f;
    for (int c = tid; c < C; c += 256) { float d = xr[c] - mean; sq += d * d; }
    red[tid] = sq; __syncthreads();
    for (int o = 128; o > 0; o >>= 1) {
        if (tid < o) red[tid] += red[tid + o];
        __syncthreads();
    }
    const float rstd = rsqrtf(red[0] / (float)C + 1e-5f);

    for (int c = tid; c < C; c += 256)
        out[(size_t)t * C + c] =
            __float2bfloat16((xr[c] - mean) * rstd * g[c] + b[c]);
}

// ---------------------------------------------------------------------------
// WMMA GEMM: C[M,N](f32 or bf16) = act( A_bf16[M,K] @ B_f32[K,N] + bias + resid )
// 128x128x32 tile, 256 threads = 8 wave32s; each wave: 2x4 grid of 16x16 tiles.
// A tile moved by TDM (wave 0 issues; D# pad gives 80B LDS row stride).
// M,N,K multiples of 128/128/32 (true for all GEMMs here).
// ---------------------------------------------------------------------------
#define BM 128
#define BN 128
#define BK 32
#define BKP 40  // LDS row stride in halves (80B = 64B row + 16B TDM pad)

template <bool OUT_BF16, bool RELU, bool BIAS, bool RESID>
__global__ __launch_bounds__(256)
void gemm_wmma_kernel(const __hip_bfloat16* __restrict__ A,
                      const float* __restrict__ B,
                      const float* __restrict__ bias,
                      const float* __restrict__ resid,
                      float* __restrict__ Cf,
                      __hip_bfloat16* __restrict__ Cb,
                      int M, int N, int K) {
    __shared__ __bf16 As[BM][BKP];  // [m][k], written by TDM
    __shared__ __bf16 Bs[BN][BKP];  // [n][k] (transposed stage => contiguous K)

    const int tid    = threadIdx.x;
    const int lane   = tid & 31;
    const int wave   = tid >> 5;   // 0..7
    const int wm     = wave & 3;   // 32-row slab
    const int wn     = wave >> 2;  // 64-col slab
    const int lanelo = lane & 15;
    const int lhi    = lane >> 4;

    const int m0 = blockIdx.y * BM;
    const int n0 = blockIdx.x * BN;

    const unsigned as_lds = (unsigned)(size_t)(&As[0][0]);  // LDS byte offset

    v8f acc[2][4];
#pragma unroll
    for (int i = 0; i < 2; i++)
#pragma unroll
        for (int j = 0; j < 4; j++) acc[i][j] = v8f{};

    // B stage: thread -> 4x4 f32 sub-tile (kq rows x nq cols)
    const int nq = (tid & 31) * 4;  // 0..124
    const int kq = (tid >> 5) * 4;  // 0..28

    for (int kk = 0; kk < K; kk += BK) {
        // ---- A tile via Tensor Data Mover (one wave issues for the WG) ----
        if (wave == 0) {
            // tile: 32 halves x 128 rows, row stride K halves,
            // LDS pad: every 16 DWORDs (64B row) add 4 DWORDs (16B) -> BKP.
            tdm_load_2d_bf16(as_lds, (const void*)(A + (size_t)m0 * K + kk),
                             BK, BM, (unsigned long long)K,
                             /*interval 16dw*/ 3, /*amount 4dw*/ 3, true);
        }
        // ---- B tile: 4x4 register transpose, f32 -> packed bf16 ----
        {
            const float* bp0 = B + (size_t)(kk + kq) * N + n0 + nq;
            float4 f0 = *(const float4*)(bp0);
            float4 f1 = *(const float4*)(bp0 + N);
            float4 f2 = *(const float4*)(bp0 + 2 * (size_t)N);
            float4 f3 = *(const float4*)(bp0 + 3 * (size_t)N);
            float a0[4] = {f0.x, f0.y, f0.z, f0.w};
            float a1[4] = {f1.x, f1.y, f1.z, f1.w};
            float a2[4] = {f2.x, f2.y, f2.z, f2.w};
            float a3[4] = {f3.x, f3.y, f3.z, f3.w};
#pragma unroll
            for (int n = 0; n < 4; n++) {
                uint2 wv;
                wv.x = pack_bf2(a0[n], a1[n]);  // k, k+1
                wv.y = pack_bf2(a2[n], a3[n]);  // k+2, k+3
                *(uint2*)(&Bs[nq + n][kq]) = wv;  // 8B contiguous in K
            }
        }
        if (kk + BK < K)  // gfx1250 global_prefetch_b8 for next B tile
            __builtin_prefetch(B + (size_t)(kk + BK + kq) * N + n0 + nq, 0, 1);

        if (wave == 0) __builtin_amdgcn_s_wait_tensorcnt(0);
        __syncthreads();

        // ---- fragments (ISA 16-bit A/B layout: lanes 0-15 hold K halves
        //      {0..7,16..23}, lanes 16-31 hold {8..15,24..31}) ----
        union Frag { uint4 u[2]; v16bf v; };
        Frag afr[2], bfr[4];
#pragma unroll
        for (int mt = 0; mt < 2; mt++) {
            const __bf16* p = &As[wm * 32 + mt * 16 + lanelo][lhi * 8];
            afr[mt].u[0] = *(const uint4*)(p);
            afr[mt].u[1] = *(const uint4*)(p + 16);
        }
#pragma unroll
        for (int nt = 0; nt < 4; nt++) {
            const __bf16* p = &Bs[wn * 64 + nt * 16 + lanelo][lhi * 8];
            bfr[nt].u[0] = *(const uint4*)(p);
            bfr[nt].u[1] = *(const uint4*)(p + 16);
        }

        // ---- 8 WMMAs per wave per K-step (EXEC all-ones here) ----
#pragma unroll
        for (int mt = 0; mt < 2; mt++)
#pragma unroll
            for (int nt = 0; nt < 4; nt++)
                acc[mt][nt] = __builtin_amdgcn_wmma_f32_16x16x32_bf16(
                    false, afr[mt].v, false, bfr[nt].v,
                    (short)0, acc[mt][nt], false, false);
        __syncthreads();
    }

    // ---- epilogue: D layout lane l -> (m = (l>=16)*8 + r, n = l&15) ----
#pragma unroll
    for (int mt = 0; mt < 2; mt++) {
#pragma unroll
        for (int nt = 0; nt < 4; nt++) {
            const int col = n0 + wn * 64 + nt * 16 + lanelo;
            float bv = 0.f;
            if (BIAS) bv = bias[col];
#pragma unroll
            for (int r = 0; r < 8; r++) {
                const int row = m0 + wm * 32 + mt * 16 + lhi * 8 + r;
                float v = acc[mt][nt][r] + bv;
                if (RESID) v += resid[(size_t)row * N + col];
                if (RELU)  v = v > 0.f ? v : 0.f;
                if (OUT_BF16) Cb[(size_t)row * N + col] = __float2bfloat16(v);
                else          Cf[(size_t)row * N + col] = v;
            }
        }
    }
}

// ---------------------------------------------------------------------------
// Causal attention, one block per (t, h, b). Dh = 64, T <= 1024. ~7% of FLOPs.
// ---------------------------------------------------------------------------
__global__ __launch_bounds__(256)
void attention_kernel(const float* __restrict__ q,
                      const float* __restrict__ k,
                      const float* __restrict__ v,
                      __hip_bfloat16* __restrict__ o,
                      int T, int C, int Dh) {
    const int t = blockIdx.x, h = blockIdx.y, b = blockIdx.z;
    const int tid = threadIdx.x;
    __shared__ float qrow[64];
    __shared__ float sc[1024];
    __shared__ float red[256];
    __shared__ float opart[4][64];

    const size_t base = (size_t)b * T * C + (size_t)h * Dh;

    if (tid < Dh) qrow[tid] = q[base + (size_t)t * C + tid];
    __syncthreads();

    float lmax = -INFINITY;
    for (int s = tid; s <= t; s += 256) {
        const float* kr = k + base + (size_t)s * C;
        float d = 0.f;
#pragma unroll 8
        for (int e = 0; e < 64; e++) d += qrow[e] * kr[e];
        d *= 0.125f;  // 1/sqrt(64)
        sc[s] = d;
        lmax = fmaxf(lmax, d);
    }
    red[tid] = lmax; __syncthreads();
    for (int off = 128; off > 0; off >>= 1) {
        if (tid < off) red[tid] = fmaxf(red[tid], red[tid + off]);
        __syncthreads();
    }
    const float mx = red[0]; __syncthreads();

    float lsum = 0.f;
    for (int s = tid; s <= t; s += 256) {
        float e = __expf(sc[s] - mx);
        sc[s] = e;
        lsum += e;
    }
    red[tid] = lsum; __syncthreads();
    for (int off = 128; off > 0; off >>= 1) {
        if (tid < off) red[tid] += red[tid + off];
        __syncthreads();
    }
    const float inv = 1.0f / red[0]; __syncthreads();

    const int dd = tid & 63;
    const int ch = tid >> 6;
    float a = 0.f;
    for (int s = ch; s <= t; s += 4)
        a += sc[s] * v[base + (size_t)s * C + dd];
    opart[ch][dd] = a;
    __syncthreads();
    if (ch == 0) {
        float r = (opart[0][dd] + opart[1][dd] + opart[2][dd] + opart[3][dd]) * inv;
        o[base + (size_t)t * C + dd] = __float2bfloat16(r);
    }
}

// ---------------------------------------------------------------------------
// Orchestration
// ---------------------------------------------------------------------------
extern "C" void kernel_launch(void* const* d_in, const int* in_sizes, int n_in,
                              void* d_out, int out_size, void* d_ws, size_t ws_size,
                              hipStream_t stream) {
    constexpr int Vv = 32000, C = 768, H = 12, Dh = 64, L = 6, Bb = 2, Tt = 1024;
    constexpr int M = Bb * Tt;  // 2048 tokens
    constexpr int F = 4 * C;    // 3072

    const int*   x_indices = (const int*)  d_in[0];
    const float* tok_emb   = (const float*)d_in[1];
    const float* pos_emb   = (const float*)d_in[2];
    const float* ln1_g     = (const float*)d_in[3];
    const float* ln1_b     = (const float*)d_in[4];
    const float* wq        = (const float*)d_in[5];
    const float* wk        = (const float*)d_in[6];
    const float* wv        = (const float*)d_in[7];
    const float* wo        = (const float*)d_in[8];
    const float* bo        = (const float*)d_in[9];
    const float* ln2_g     = (const float*)d_in[10];
    const float* ln2_b     = (const float*)d_in[11];
    const float* w1        = (const float*)d_in[12];
    const float* b1        = (const float*)d_in[13];
    const float* w2        = (const float*)d_in[14];
    const float* b2        = (const float*)d_in[15];
    const float* lnf_g     = (const float*)d_in[16];
    const float* lnf_b     = (const float*)d_in[17];
    const float* wp        = (const float*)d_in[18];
    const float* bp        = (const float*)d_in[19];
    float* logits          = (float*)d_out;

    uint8_t* w = (uint8_t*)d_ws;
    auto take = [&](size_t bytes) {
        uint8_t* p = w;
        w += (bytes + 255) & ~(size_t)255;
        return p;
    };
    float*          x  = (float*)          take((size_t)M * C * 4);
    __hip_bfloat16* hb = (__hip_bfloat16*) take((size_t)M * C * 2);
    float*          qb = (float*)          take((size_t)M * C * 4);
    float*          kb = (float*)          take((size_t)M * C * 4);
    float*          vb = (float*)          take((size_t)M * C * 4);
    __hip_bfloat16* ob = (__hip_bfloat16*) take((size_t)M * C * 2);
    __hip_bfloat16* ub = (__hip_bfloat16*) take((size_t)M * F * 2);
    (void)ws_size; (void)in_sizes; (void)n_in; (void)out_size;

    const dim3 blk(256);
    const dim3 gC(C / BN, M / BM);   // 6  x 16
    const dim3 gF(F / BN, M / BM);   // 24 x 16
    const dim3 gV(Vv / BN, M / BM);  // 250 x 16
    const dim3 gAtt(Tt, H, Bb);

    {
        long total = (long)M * C;
        embed_kernel<<<(unsigned)((total + 255) / 256), blk, 0, stream>>>(
            x_indices, tok_emb, pos_emb, x, Tt, C, total);
    }

    for (int l = 0; l < L; l++) {
        const float* wq_l = wq + (size_t)l * C * C;
        const float* wk_l = wk + (size_t)l * C * C;
        const float* wv_l = wv + (size_t)l * C * C;
        const float* wo_l = wo + (size_t)l * C * C;
        const float* w1_l = w1 + (size_t)l * C * F;
        const float* w2_l = w2 + (size_t)l * F * C;

        layernorm_bf16_kernel<<<M, blk, 0, stream>>>(x, ln1_g + l * C, ln1_b + l * C, hb, C);

        gemm_wmma_kernel<false, false, false, false><<<gC, blk, 0, stream>>>(
            hb, wq_l, nullptr, nullptr, qb, nullptr, M, C, C);
        gemm_wmma_kernel<false, false, false, false><<<gC, blk, 0, stream>>>(
            hb, wk_l, nullptr, nullptr, kb, nullptr, M, C, C);
        gemm_wmma_kernel<false, false, false, false><<<gC, blk, 0, stream>>>(
            hb, wv_l, nullptr, nullptr, vb, nullptr, M, C, C);

        attention_kernel<<<gAtt, blk, 0, stream>>>(qb, kb, vb, ob, Tt, C, Dh);

        gemm_wmma_kernel<false, false, true, true><<<gC, blk, 0, stream>>>(
            ob, wo_l, bo + l * C, x, x, nullptr, M, C, C);

        layernorm_bf16_kernel<<<M, blk, 0, stream>>>(x, ln2_g + l * C, ln2_b + l * C, hb, C);

        gemm_wmma_kernel<true, true, true, false><<<gF, blk, 0, stream>>>(
            hb, w1_l, b1 + (size_t)l * F, nullptr, nullptr, ub, M, F, C);

        gemm_wmma_kernel<false, false, true, true><<<gC, blk, 0, stream>>>(
            ub, w2_l, b2 + l * C, x, x, nullptr, M, C, F);
    }

    layernorm_bf16_kernel<<<M, blk, 0, stream>>>(x, lnf_g, lnf_b, hb, C);
    gemm_wmma_kernel<false, false, true, false><<<gV, blk, 0, stream>>>(
        hb, wp, bp, nullptr, logits, nullptr, M, Vv, C);
}
